// LSTM_Attention_8735963480318
// MI455X (gfx1250) — matrix operand order repaired and checked by reading, not compiled, not run on hardware
//
#include <hip/hip_runtime.h>
#include <hip/hip_bf16.h>
#include <math.h>

// Problem constants (match reference)
#define SDIM 512
#define NDIM 64
#define IDIM 256
#define HDIM 1024
#define TDIM 128

typedef __bf16 bf16;
typedef __attribute__((ext_vector_type(16))) __bf16 v16bf;
typedef __attribute__((ext_vector_type(8)))  __bf16 v8bf;
typedef __attribute__((ext_vector_type(8)))  float  v8f;

// ---------- scalar bf16 helpers (bit-exact RNE, no reliance on __bf16 arith) ----------
__device__ __forceinline__ bf16 f2bf(float x) {
  unsigned int u = __float_as_uint(x);
  unsigned int r = (u + 0x7fffu + ((u >> 16) & 1u)) >> 16;
  unsigned short s = (unsigned short)r;
  bf16 b;
  __builtin_memcpy(&b, &s, 2);
  return b;
}
__device__ __forceinline__ float bf2f(bf16 x) {
  unsigned short s;
  __builtin_memcpy(&s, &x, 2);
  unsigned int u = ((unsigned int)s) << 16;
  return __uint_as_float(u);
}
__device__ __forceinline__ float sigmoidf_(float x) { return 1.0f / (1.0f + expf(-x)); }

// ---------- WMMA tile loaders (bf16 16x16x32, wave32; contiguous k order 16*(lane/16)+s, same for A and B) ----
// A (16x32, row-major source, rows = m_base..m_base+15):
//   lanes 0-15:  row m_base+l, K = k0..k0+15 in v[0..15]
//   lanes 16-31: row m_base+(l-16), K = k0+16..k0+31
__device__ __forceinline__ v16bf load_a_tile(const bf16* A, int lda, int m_base, int k0, int lane) {
  const int row = m_base + (lane & 15);
  const int kk  = k0 + ((lane >> 4) << 4);
  const bf16* p = A + (size_t)row * (size_t)lda + kk;
  union { v16bf v; v8bf h[2]; } u;
  u.h[0] = *(const v8bf*)(p);
  u.h[1] = *(const v8bf*)(p + 8);
  return u.v;
}
// B = W^T where W is (Ncols x K) row-major: B[k][j] = W[j][k].
//   lane 0-15:  col n_base+l, K = k0..k0+15  -> 16 contiguous bf16 of row j of W
//   lane 16-31: col n_base+(l-16), K = k0+16..k0+31
__device__ __forceinline__ v16bf load_bT_tile(const bf16* W, int ldw, int n_base, int k0, int lane) {
  const int jrow = n_base + (lane & 15);
  const int kk   = k0 + ((lane >> 4) << 4);
  return *(const v16bf*)(W + (size_t)jrow * (size_t)ldw + kk);
}
__device__ __forceinline__ v8f wmma_bf16(v16bf a, v16bf b, v8f c) {
  return __builtin_amdgcn_wmma_f32_16x16x32_bf16(false, a, false, b, (short)0, c, false, false);
}
__device__ __forceinline__ v8f zero_acc() {
  v8f z;
#pragma unroll
  for (int i = 0; i < 8; ++i) z[i] = 0.0f;
  return z;
}

// ---------------------------- utility kernels ----------------------------
__global__ void f2bf_kernel(const float* __restrict__ in, bf16* __restrict__ out, int n) {
  int i = blockIdx.x * 256 + threadIdx.x;
  if (i < n) out[i] = f2bf(in[i]);
}
__global__ void zero_f32_kernel(float* __restrict__ p, int n) {
  int i = blockIdx.x * 256 + threadIdx.x;
  if (i < n) p[i] = 0.0f;
}
__global__ void zero_bf_kernel(bf16* __restrict__ p, int n) {
  int i = blockIdx.x * 256 + threadIdx.x;
  if (i < n) p[i] = f2bf(0.0f);
}

// ------------------- fused LSTM step (encoder & decoder) -------------------
// gates(N x 4H) = bias + xin(N x I) @ Wih^T + hin(N x H) @ Whh^T, then cell update.
// grid.x = H/16 (hidden column tiles), block = 128 threads = 4 waves.
// WAVE = GATE: each wave streams its gate's weight rows exactly once (no intra-
// block redundancy on the 8MB Whh stream) and re-reads the small hot h operand.
// Gate tiles are exchanged through LDS for the fused pointwise cell update.
__global__ __launch_bounds__(128)
void lstm_step_kernel(const bf16* __restrict__ xin,   // (N, I)
                      const bf16* __restrict__ Wih,   // (4H, I)
                      const bf16* __restrict__ Whh,   // (4H, H)
                      const float* __restrict__ bias, // (4H)
                      const bf16* __restrict__ hin,   // (N, H)
                      float* __restrict__ c,          // (N, H) in/out
                      bf16* __restrict__ hout,        // (N, H)
                      bf16* __restrict__ oute)        // (N, H) or nullptr
{
  __shared__ float gsm[4 * NDIM * 16];                // [gate][row 0..63][col 0..15], 16KB

  const int lane   = threadIdx.x & 31;
  const int g      = threadIdx.x >> 5;                // wave index == gate index
  const int j_base = blockIdx.x * 16;

  v8f acc[4];                                         // 4 batch M-tiles (rows 0..63)
#pragma unroll
  for (int mt = 0; mt < 4; ++mt) acc[mt] = zero_acc();

  // recurrent contribution: hin @ Whh^T  (K = HDIM); B loaded once per k-step
#pragma unroll 2
  for (int k0 = 0; k0 < HDIM; k0 += 32) {
    v16bf b = load_bT_tile(Whh, HDIM, g * HDIM + j_base, k0, lane);
#pragma unroll
    for (int mt = 0; mt < 4; ++mt) {
      v16bf a = load_a_tile(hin, HDIM, mt * 16, k0, lane);
      acc[mt] = wmma_bf16(a, b, acc[mt]);
    }
  }
  // input contribution: xin @ Wih^T  (K = IDIM)
#pragma unroll 2
  for (int k0 = 0; k0 < IDIM; k0 += 32) {
    v16bf b = load_bT_tile(Wih, IDIM, g * HDIM + j_base, k0, lane);
#pragma unroll
    for (int mt = 0; mt < 4; ++mt) {
      v16bf a = load_a_tile(xin, IDIM, mt * 16, k0, lane);
      acc[mt] = wmma_bf16(a, b, acc[mt]);
    }
  }

  // spill this wave's gate tiles to LDS: D elem r <-> row r/(r+8) by lane half
  const int colL = lane & 15;
  const int rhi  = (lane >> 4) << 3;
#pragma unroll
  for (int mt = 0; mt < 4; ++mt) {
#pragma unroll
    for (int r = 0; r < 8; ++r) {
      const int row = mt * 16 + r + rhi;
      gsm[(g * NDIM + row) * 16 + colL] = acc[mt][r];
    }
  }
  __syncthreads();

  // fused pointwise cell update: 128 threads cover 64 rows x 16 cols
  for (int e = threadIdx.x; e < NDIM * 16; e += 128) {
    const int row = e >> 4;
    const int c16 = e & 15;
    const int col = j_base + c16;
    float iv = sigmoidf_(gsm[(0 * NDIM + row) * 16 + c16] + bias[0 * HDIM + col]);
    float fv = sigmoidf_(gsm[(1 * NDIM + row) * 16 + c16] + bias[1 * HDIM + col]);
    float gv = tanhf    (gsm[(2 * NDIM + row) * 16 + c16] + bias[2 * HDIM + col]);
    float ov = sigmoidf_(gsm[(3 * NDIM + row) * 16 + c16] + bias[3 * HDIM + col]);
    const int idx = row * HDIM + col;
    float cn = fv * c[idx] + iv * gv;
    float hn = ov * tanhf(cn);
    c[idx] = cn;
    bf16 hb = f2bf(hn);
    hout[idx] = hb;
    if (oute) oute[idx] = hb;
  }
}

// ------------------- generic GEMM: C = A(MxK) @ W(NxK)^T + bias -------------------
// grid = (ceil(M/64), N/16), block = 128 = 4 waves, wave -> 16x16 output tile.
// Wave = M-tile here: the streamed operand (A) is distinct per wave; the shared
// B operand (W, <=512KB) is L2-resident, so its intra-block re-read is cheap.
__global__ __launch_bounds__(128)
void gemm_bf16_kernel(const bf16* __restrict__ A, const bf16* __restrict__ W,
                      const float* __restrict__ bias,
                      float* __restrict__ Cout, bf16* __restrict__ Cbf,
                      int M, int Ncols, int K)
{
  const int lane   = threadIdx.x & 31;
  const int wave   = threadIdx.x >> 5;
  const int m_base = (blockIdx.x * 4 + wave) * 16;
  const int n_base = blockIdx.y * 16;
  if (m_base >= M) return;  // wave-uniform

  v8f acc = zero_acc();
#pragma unroll 2
  for (int k0 = 0; k0 < K; k0 += 32) {
    v16bf a = load_a_tile(A, K, m_base, k0, lane);
    v16bf b = load_bT_tile(W, K, n_base, k0, lane);
    acc = wmma_bf16(a, b, acc);
  }
  const int col = n_base + (lane & 15);
  const float bv = bias ? bias[col] : 0.0f;
#pragma unroll
  for (int r = 0; r < 8; ++r) {
    const int row = m_base + r + ((lane >> 4) << 3);
    float v = acc[r] + bv;
    if (Cout) Cout[(size_t)row * Ncols + col] = v;
    if (Cbf)  Cbf [(size_t)row * Ncols + col] = f2bf(v);
  }
}

// ------------------- attention scores: one wave per (s,n) -------------------
// scores[s,n] = b3 + sum_i w3[i]*tanh(pre2[s,n,i] + hw1[n,i])
__global__ __launch_bounds__(128)
void scores_kernel(const float* __restrict__ pre2, const float* __restrict__ hw1,
                   const float* __restrict__ w3, const float* __restrict__ b3,
                   float* __restrict__ scores)
{
  const int lane = threadIdx.x & 31;
  const int wave = threadIdx.x >> 5;
  const int sn   = blockIdx.x * 4 + wave;           // sn = s*N + n
  const int n    = sn & (NDIM - 1);
  const float* p = pre2 + (size_t)sn * IDIM;
  const float* q = hw1  + (size_t)n  * IDIM;
  float acc = 0.0f;
  for (int i = lane; i < IDIM; i += 32)
    acc += w3[i] * tanhf(p[i] + q[i]);
#pragma unroll
  for (int off = 16; off > 0; off >>= 1)
    acc += __shfl_xor(acc, off, 32);
  if (lane == 0) scores[sn] = acc + b3[0];
}

// ------------------- softmax along S for each batch column n -------------------
__global__ __launch_bounds__(256)
void softmax_kernel(const float* __restrict__ scores, float* __restrict__ alpha)
{
  __shared__ float red[256];
  const int n = blockIdx.x, tid = threadIdx.x;
  float m = -1e30f;
  for (int s = tid; s < SDIM; s += 256) m = fmaxf(m, scores[s * NDIM + n]);
  red[tid] = m; __syncthreads();
  for (int w = 128; w > 0; w >>= 1) { if (tid < w) red[tid] = fmaxf(red[tid], red[tid + w]); __syncthreads(); }
  m = red[0]; __syncthreads();
  float sum = 0.0f;
  for (int s = tid; s < SDIM; s += 256) {
    float e = expf(scores[s * NDIM + n] - m);
    alpha[s * NDIM + n] = e; sum += e;
  }
  red[tid] = sum; __syncthreads();
  for (int w = 128; w > 0; w >>= 1) { if (tid < w) red[tid] += red[tid + w]; __syncthreads(); }
  const float inv = 1.0f / red[0];
  for (int s = tid; s < SDIM; s += 256) alpha[s * NDIM + n] *= inv;
}

// ------------- context: ctx[n,h] = sum_s alpha[s,n]*out_e[s,n,h] -------------
// One block per n; alpha column staged in LDS; v8bf (16B) vector loads of out_e.
__global__ __launch_bounds__(128)
void ctx_kernel(const float* __restrict__ alpha, const bf16* __restrict__ oute,
                bf16* __restrict__ ctx_bf)
{
  __shared__ float al[SDIM];
  const int n = blockIdx.x;
  for (int s = threadIdx.x; s < SDIM; s += 128) al[s] = alpha[s * NDIM + n];
  __syncthreads();

  const int h0 = threadIdx.x * 8;                    // 128 threads * 8 = HDIM
  float acc[8];
#pragma unroll
  for (int i = 0; i < 8; ++i) acc[i] = 0.0f;
  for (int s = 0; s < SDIM; ++s) {
    v8bf v = *(const v8bf*)(oute + ((size_t)(s * NDIM + n)) * HDIM + h0);
    const float a = al[s];
#pragma unroll
    for (int i = 0; i < 8; ++i) acc[i] += a * bf2f(v[i]);
  }
  v8bf o;
#pragma unroll
  for (int i = 0; i < 8; ++i) o[i] = f2bf(acc[i]);
  *(v8bf*)(ctx_bf + n * HDIM + h0) = o;
}

// ============================== host launch ==============================
extern "C" void kernel_launch(void* const* d_in, const int* in_sizes, int n_in,
                              void* d_out, int out_size, void* d_ws, size_t ws_size,
                              hipStream_t stream) {
  (void)in_sizes; (void)n_in; (void)out_size; (void)ws_size;

  const float* x     = (const float*)d_in[0];
  const float* We_ih = (const float*)d_in[2];
  const float* We_hh = (const float*)d_in[3];
  const float* be    = (const float*)d_in[4];
  const float* Wd_ih = (const float*)d_in[5];
  const float* Wd_hh = (const float*)d_in[6];
  const float* bd    = (const float*)d_in[7];
  const float* W1    = (const float*)d_in[8];
  const float* b1    = (const float*)d_in[9];
  const float* W2    = (const float*)d_in[10];
  const float* b2    = (const float*)d_in[11];
  const float* w3    = (const float*)d_in[12];
  const float* b3    = (const float*)d_in[13];
  const float* Wl    = (const float*)d_in[14];
  const float* bl    = (const float*)d_in[15];
  float* out = (float*)d_out;

  // -------- workspace carve-out (~142 MB total) --------
  char* wp = (char*)d_ws;
  auto carve = [&](size_t bytes) -> void* {
    void* r = (void*)wp;
    wp += (bytes + 255) & ~(size_t)255;
    return r;
  };
  bf16* x_bf    = (bf16*)carve((size_t)SDIM * NDIM * IDIM * 2);
  bf16* Weih_bf = (bf16*)carve((size_t)4 * HDIM * IDIM * 2);
  bf16* Wehh_bf = (bf16*)carve((size_t)4 * HDIM * HDIM * 2);
  bf16* Wdih_bf = (bf16*)carve((size_t)4 * HDIM * IDIM * 2);
  bf16* Wdhh_bf = (bf16*)carve((size_t)4 * HDIM * HDIM * 2);
  bf16* W1_bf   = (bf16*)carve((size_t)IDIM * HDIM * 2);
  bf16* W2_bf   = (bf16*)carve((size_t)IDIM * HDIM * 2);
  bf16* Wl_bf   = (bf16*)carve((size_t)IDIM * HDIM * 2);
  bf16* oute_bf = (bf16*)carve((size_t)SDIM * NDIM * HDIM * 2);
  float* pre2   = (float*)carve((size_t)SDIM * NDIM * IDIM * 4);
  bf16* hbuf0   = (bf16*)carve((size_t)NDIM * HDIM * 2);
  bf16* hbuf1   = (bf16*)carve((size_t)NDIM * HDIM * 2);
  float* cbuf   = (float*)carve((size_t)NDIM * HDIM * 4);
  float* hw1    = (float*)carve((size_t)NDIM * IDIM * 4);
  float* scores = (float*)carve((size_t)SDIM * NDIM * 4);
  float* alpha  = (float*)carve((size_t)SDIM * NDIM * 4);
  bf16* ctx_bf  = (bf16*)carve((size_t)NDIM * HDIM * 2);
  bf16* inpd_bf = (bf16*)carve((size_t)NDIM * IDIM * 2);

  auto cvt = [&](const float* src, bf16* dst, int n) {
    f2bf_kernel<<<(n + 255) / 256, 256, 0, stream>>>(src, dst, n);
  };
  cvt(x,     x_bf,    SDIM * NDIM * IDIM);
  cvt(We_ih, Weih_bf, 4 * HDIM * IDIM);
  cvt(We_hh, Wehh_bf, 4 * HDIM * HDIM);
  cvt(Wd_ih, Wdih_bf, 4 * HDIM * IDIM);
  cvt(Wd_hh, Wdhh_bf, 4 * HDIM * HDIM);
  cvt(W1,    W1_bf,   IDIM * HDIM);
  cvt(W2,    W2_bf,   IDIM * HDIM);
  cvt(Wl,    Wl_bf,   IDIM * HDIM);

  zero_bf_kernel <<<(NDIM * HDIM + 255) / 256, 256, 0, stream>>>(hbuf0, NDIM * HDIM);
  zero_f32_kernel<<<(NDIM * HDIM + 255) / 256, 256, 0, stream>>>(cbuf,  NDIM * HDIM);

  // -------- encoder: 512 sequential fused LSTM steps (double-buffered h) --------
  bf16* hbufs[2] = {hbuf0, hbuf1};
  for (int t = 0; t < SDIM; ++t) {
    lstm_step_kernel<<<HDIM / 16, 128, 0, stream>>>(
        x_bf + (size_t)t * NDIM * IDIM, Weih_bf, Wehh_bf, be,
        hbufs[t & 1], cbuf, hbufs[(t + 1) & 1],
        oute_bf + (size_t)t * NDIM * HDIM);
  }
  // final h_e lives in hbuf0 (t=511 writes buffer 0); c_e lives in cbuf.

  // -------- pre2 = out_e @ W2^T + b2 : (32768 x 256 x 1024) WMMA GEMM --------
  gemm_bf16_kernel<<<dim3((SDIM * NDIM) / 64, IDIM / 16), 128, 0, stream>>>(
      oute_bf, W2_bf, b2, pre2, nullptr, SDIM * NDIM, IDIM, HDIM);

  // input_d(0) = x[-1]
  hipMemcpyAsync(inpd_bf, x_bf + (size_t)(SDIM - 1) * NDIM * IDIM,
                 (size_t)NDIM * IDIM * sizeof(bf16), hipMemcpyDeviceToDevice, stream);

  // -------- decoder: 128 sequential attention+LSTM steps --------
  for (int t = 0; t < TDIM; ++t) {
    // hw1 = h_d @ W1^T + b1
    gemm_bf16_kernel<<<dim3(1, IDIM / 16), 128, 0, stream>>>(
        hbuf0, W1_bf, b1, hw1, nullptr, NDIM, IDIM, HDIM);
    // scores over (S,N)
    scores_kernel<<<(SDIM * NDIM) / 4, 128, 0, stream>>>(pre2, hw1, w3, b3, scores);
    // softmax along S
    softmax_kernel<<<NDIM, 256, 0, stream>>>(scores, alpha);
    // ctx[n,h] = sum_s alpha * out_e
    ctx_kernel<<<NDIM, 128, 0, stream>>>(alpha, oute_bf, ctx_bf);
    // decoder LSTM step: hidden = ctx, cell = cbuf -> h_new in hbuf0
    lstm_step_kernel<<<HDIM / 16, 128, 0, stream>>>(
        inpd_bf, Wdih_bf, Wdhh_bf, bd, ctx_bf, cbuf, hbuf0, nullptr);
    // out_d = h_new @ Wl^T + bl -> d_out[t] (f32) and next input (bf16)
    gemm_bf16_kernel<<<dim3(1, IDIM / 16), 128, 0, stream>>>(
        hbuf0, Wl_bf, bl, out + (size_t)t * NDIM * IDIM, inpd_bf, NDIM, IDIM, HDIM);
  }
}
